// TitanBlock_7610682048661
// MI455X (gfx1250) — matrix operand order repaired and checked by reading, not compiled, not run on hardware
//
#include <hip/hip_runtime.h>
#include <math.h>

// ---------------------------------------------------------------------------
// CDNA5 (gfx1250) wave32 transformer block:
//   - v_wmma_f32_16x16x32_bf16 for all GEMMs (f32 accumulate)
//   - Tensor Data Mover (tensor_load_to_lds + s_wait_tensorcnt) stages all
//     GEMM tiles HBM->LDS; f32 weights converted to bf16 by a cooperative
//     LDS->LDS pass
//   - flash attention with online softmax; MoE top-6 tile skipping
// ---------------------------------------------------------------------------

typedef __attribute__((ext_vector_type(16))) __bf16 v16bf;
typedef __attribute__((ext_vector_type(8)))  float  v8f;
typedef __attribute__((ext_vector_type(4)))  unsigned int v4u;
typedef __attribute__((ext_vector_type(8)))  int v8i;
typedef __attribute__((ext_vector_type(4)))  int v4i;

#define WMMA_BF16(A, B, C) \
  __builtin_amdgcn_wmma_f32_16x16x32_bf16(false, (A), false, (B), (short)0, (C), false, false)

union BFrag { v16bf v; unsigned short u[16]; };

__device__ __forceinline__ unsigned short f2bf(float f) {
  union { float f; unsigned int u; } c; c.f = f;
  unsigned int u = c.u;
  u += 0x7FFFu + ((u >> 16) & 1u);            // round-to-nearest-even
  return (unsigned short)(u >> 16);
}
__device__ __forceinline__ float bf2f(unsigned short h) {
  union { unsigned int u; float f; } c; c.u = ((unsigned int)h) << 16;
  return c.f;
}

// A fragment (16x32 bf16): lane = row, interleaved K halves per ISA table.
__device__ __forceinline__ v16bf fragA(const unsigned short* p, int ld, int koff) {
  int lane = threadIdx.x & 31;
  int r = lane & 15;
  int kh = (lane >> 4) * 8;
  const unsigned short* q = p + (size_t)r * ld + koff;
  BFrag f;
#pragma unroll
  for (int i = 0; i < 8; ++i) { f.u[i] = q[kh + i]; f.u[i + 8] = q[16 + kh + i]; }
  return f.v;
}
// B fragment (32x16 bf16) from "row per output-column over K" storage.
__device__ __forceinline__ v16bf fragB_row(const unsigned short* p, int ld, int koff) {
  int lane = threadIdx.x & 31;
  int n = lane & 15;
  int kb = (lane >> 4) * 8;
  const unsigned short* q = p + (size_t)n * ld + koff + kb;
  BFrag f;
#pragma unroll
  for (int i = 0; i < 8; ++i) { f.u[i] = q[i]; f.u[i + 8] = q[16 + i]; }
  return f.v;
}

// ---------------------------------------------------------------------------
// Tensor Data Mover: 2-D tile load, D# built per ISA 08_async_tensor §8.3-8.5.
// dsz_code: 0=1B,1=2B,2=4B.  ld / tile dims in data_size units.
// amdgpu-toolchain (clang-23) 6-arg builtin:
//   (uint32x4 g0, int32x8 g1, int32x4 g2, int32x4 g3, int32x8 g4, i32 cpol)
__device__ __forceinline__ void tdm_load_2d(unsigned lds_off, const void* gptr,
                                            unsigned dsz_code, unsigned tile_cols,
                                            unsigned tile_rows, unsigned ld) {
  unsigned long long ga = (unsigned long long)(size_t)gptr;
  v4u g0;
  g0[0] = 1u;                                           // count=1, user mode
  g0[1] = lds_off;                                      // lds_addr (bytes)
  g0[2] = (unsigned)ga;                                 // global_addr[31:0]
  g0[3] = (unsigned)((ga >> 32) & 0x01FFFFFFu) | 0x80000000u; // addr[56:32] | type=2
  unsigned dim1 = 1u << 24;                             // large: tiles always in-bounds
  v8i g1;
  g1[0] = (int)(dsz_code << 16);                        // mask=0, data_size
  g1[1] = (int)((ld & 0xFFFFu) << 16);                  // tensor_dim0[15:0]
  g1[2] = (int)((ld >> 16) | ((dim1 & 0xFFFFu) << 16)); // dim0[31:16] | dim1[15:0]
  g1[3] = (int)((dim1 >> 16) | (tile_cols << 16));      // dim1[31:16] | tile_dim0
  g1[4] = (int)tile_rows;                               // tile_dim1 | tile_dim2=0
  g1[5] = (int)ld;                                      // tensor_dim0_stride[31:0]
  g1[6] = 0;                                            // stride0 hi | stride1 lo
  g1[7] = 0;
  v4i gz4 = {0, 0, 0, 0};
  v8i gz8 = {0, 0, 0, 0, 0, 0, 0, 0};
  __builtin_amdgcn_tensor_load_to_lds(g0, g1, gz4, gz4, gz8, 0);
}

// ---------------------------------------------------------------------------
// Shared GEMM mainloop: C[64x64] tile, K-step 32.
// A: bf16 [*, lda]; B: f32 [*, ldb] (rows = output columns).
// TDM stages both tiles; wave 0 issues, all waves s_wait_tensorcnt + barrier.
__device__ __forceinline__ void gemm_mainloop(const unsigned short* A, int lda,
                                              const float* B, int ldb, int K,
                                              int m0, int n0,
                                              unsigned short* At, float* Bf,
                                              unsigned short* Bt,
                                              v8f& acc0, v8f& acc1) {
  int wv = threadIdx.x >> 5;
  int mt = (wv & 3) * 16, nt = (wv >> 2) * 32;
  unsigned at_off = (unsigned)(size_t)(void*)At;
  unsigned bf_off = (unsigned)(size_t)(void*)Bf;
  for (int k0 = 0; k0 < K; k0 += 32) {
    if (wv == 0) {
      tdm_load_2d(at_off, A + (size_t)m0 * lda + k0, 1u, 32u, 64u, (unsigned)lda);
      tdm_load_2d(bf_off, B + (size_t)n0 * ldb + k0, 2u, 32u, 64u, (unsigned)ldb);
    }
    __builtin_amdgcn_s_wait_tensorcnt(0);
    __syncthreads();
    if (k0 + 32 < K)                                    // warm L2 for next tile
      __builtin_prefetch(B + (size_t)n0 * ldb + k0 + 32, 0, 1);
    { int idx = threadIdx.x * 8;                        // f32 -> bf16 LDS pass
#pragma unroll
      for (int i = 0; i < 8; ++i) Bt[idx + i] = f2bf(Bf[idx + i]); }
    __syncthreads();
    v16bf a = fragA(At + mt * 32, 32, 0);
    acc0 = WMMA_BF16(a, fragB_row(Bt + nt * 32, 32, 0), acc0);
    acc1 = WMMA_BF16(a, fragB_row(Bt + (nt + 16) * 32, 32, 0), acc1);
    __syncthreads();
  }
}

// ---------------------------------------------------------------------------
__global__ __launch_bounds__(256) void k_layernorm(const float* __restrict__ x,
                                                   const float* __restrict__ g,
                                                   const float* __restrict__ b,
                                                   unsigned short* __restrict__ out) {
  int row = blockIdx.x;
  const float* xr = x + (size_t)row * 1024;
  float s = 0.f, s2 = 0.f;
  for (int i = threadIdx.x; i < 1024; i += 256) { float v = xr[i]; s += v; s2 += v * v; }
#pragma unroll
  for (int o = 16; o; o >>= 1) { s += __shfl_xor(s, o, 32); s2 += __shfl_xor(s2, o, 32); }
  __shared__ float ps[8], ps2[8];
  __shared__ float mean_s, inv_s;
  int wv = threadIdx.x >> 5, lane = threadIdx.x & 31;
  if (lane == 0) { ps[wv] = s; ps2[wv] = s2; }
  __syncthreads();
  if (threadIdx.x == 0) {
    float t = 0.f, t2 = 0.f;
    for (int i = 0; i < 8; ++i) { t += ps[i]; t2 += ps2[i]; }
    float m = t * (1.f / 1024.f);
    float var = t2 * (1.f / 1024.f) - m * m;
    mean_s = m; inv_s = rsqrtf(var + 1e-5f);
  }
  __syncthreads();
  float m = mean_s, inv = inv_s;
  unsigned short* orow = out + (size_t)row * 1024;
  for (int i = threadIdx.x; i < 1024; i += 256)
    orow[i] = f2bf((xr[i] - m) * inv * g[i] + b[i]);
}

// ---------------------------------------------------------------------------
// qkv = h1 @ w_qkv^T + b_qkv -> per-head bf16 q (pre-scaled 1/8), k, v.
// n-block of 64 divides 1024 => q/k/v selection is uniform per block.
__global__ __launch_bounds__(256) void k_gemm_qkv(const unsigned short* __restrict__ h1,
                                                  const float* __restrict__ w,
                                                  const float* __restrict__ bias,
                                                  unsigned short* __restrict__ qh,
                                                  unsigned short* __restrict__ kh,
                                                  unsigned short* __restrict__ vh) {
  __shared__ __align__(16) unsigned short At[64 * 32];
  __shared__ __align__(16) float          Bf[64 * 32];
  __shared__ __align__(16) unsigned short Bt[64 * 32];
  int m0 = blockIdx.y * 64, n0 = blockIdx.x * 64;
  v8f acc0 = {}, acc1 = {};
  gemm_mainloop(h1, 1024, w, 1024, 1024, m0, n0, At, Bf, Bt, acc0, acc1);

  int wv = threadIdx.x >> 5, lane = threadIdx.x & 31;
  int mt = (wv & 3) * 16, nt = (wv >> 2) * 32;
  int half = lane >> 4, cn = lane & 15;
  int which = n0 >> 10;                                    // 0=q,1=k,2=v (uniform)
  unsigned short* dst = (which == 0) ? qh : ((which == 1) ? kh : vh);
  float qscale = (which == 0) ? 0.125f : 1.0f;             // 1/sqrt(64) folded into q
  int d0 = (n0 & 1023) + nt;
  int bb = m0 >> 10;
  int sbase = (m0 & 1023) + mt + 8 * half;
#pragma unroll
  for (int i = 0; i < 8; ++i) {
    int ss = sbase + i;
#pragma unroll
    for (int t = 0; t < 2; ++t) {
      int j = n0 + nt + t * 16 + cn;
      float v = ((t ? acc1[i] : acc0[i]) + bias[j]) * qscale;
      int d = d0 + t * 16 + cn;
      int hh = d >> 6, hd = d & 63;
      dst[((size_t)(bb * 16 + hh) * 1024 + ss) * 64 + hd] = f2bf(v);
    }
  }
}

// ---------------------------------------------------------------------------
// Flash attention: block = (b,h) x 128 query rows; wave owns 16 rows.
// V is transposed during staging so every fragment read is a b128 LDS load.
__global__ __launch_bounds__(256) void k_attn(const unsigned short* __restrict__ qh,
                                              const unsigned short* __restrict__ kh,
                                              const unsigned short* __restrict__ vh,
                                              unsigned short* __restrict__ ao) {
  __shared__ __align__(16) unsigned short kt[32 * 64];   // [token][d]
  __shared__ __align__(16) unsigned short vtt[64 * 32];  // [d][token]
  __shared__ __align__(16) unsigned short pt[8][16 * 32];
  int bh = blockIdx.y;
  int q0 = blockIdx.x * 128;
  int wv = threadIdx.x >> 5, lane = threadIdx.x & 31;
  int mrow = q0 + wv * 16;
  const unsigned short* qb = qh + (size_t)bh * 1024 * 64;
  const unsigned short* kb = kh + (size_t)bh * 1024 * 64;
  const unsigned short* vb = vh + (size_t)bh * 1024 * 64;
  v16bf qa0 = fragA(qb + (size_t)mrow * 64, 64, 0);
  v16bf qa1 = fragA(qb + (size_t)mrow * 64, 64, 32);
  v8f o0 = {}, o1 = {}, o2 = {}, o3 = {};
  float mprev[8], lsum[8];
#pragma unroll
  for (int i = 0; i < 8; ++i) { mprev[i] = -1e30f; lsum[i] = 0.f; }

  for (int c0 = 0; c0 < 1024; c0 += 32) {
    { int idx = threadIdx.x * 8; int r = idx >> 6, c = idx & 63;
      *(uint4*)(kt + idx) = *(const uint4*)(kb + (size_t)(c0 + r) * 64 + c);
      uint4 vv = *(const uint4*)(vb + (size_t)(c0 + r) * 64 + c);
      const unsigned short* pv = (const unsigned short*)&vv;
#pragma unroll
      for (int i = 0; i < 8; ++i) vtt[(c + i) * 32 + r] = pv[i]; }
    __syncthreads();
    v8f s0 = {}, s1 = {};
    s0 = WMMA_BF16(qa0, fragB_row(kt, 64, 0),  s0);
    s0 = WMMA_BF16(qa1, fragB_row(kt, 64, 32), s0);
    s1 = WMMA_BF16(qa0, fragB_row(kt + 16 * 64, 64, 0),  s1);
    s1 = WMMA_BF16(qa1, fragB_row(kt + 16 * 64, 64, 32), s1);

    unsigned short* pw = pt[wv];
    int half = lane >> 4, cn = lane & 15;
#pragma unroll
    for (int i = 0; i < 8; ++i) {
      float a = s0[i], b = s1[i];
      float mx = fmaxf(a, b);
#pragma unroll
      for (int o = 8; o; o >>= 1) mx = fmaxf(mx, __shfl_xor(mx, o, 16));
      float mnew = fmaxf(mprev[i], mx);
      float sc = __expf(mprev[i] - mnew);
      float e0 = __expf(a - mnew), e1 = __expf(b - mnew);
      float rs = e0 + e1;
#pragma unroll
      for (int o = 8; o; o >>= 1) rs += __shfl_xor(rs, o, 16);
      lsum[i] = lsum[i] * sc + rs;
      mprev[i] = mnew;
      o0[i] *= sc; o1[i] *= sc; o2[i] *= sc; o3[i] *= sc;
      int rr = i + 8 * half;
      pw[rr * 32 + cn]      = f2bf(e0);
      pw[rr * 32 + 16 + cn] = f2bf(e1);
    }
    v16bf pa = fragA(pw, 32, 0);
    o0 = WMMA_BF16(pa, fragB_row(vtt,           32, 0), o0);
    o1 = WMMA_BF16(pa, fragB_row(vtt + 16 * 32, 32, 0), o1);
    o2 = WMMA_BF16(pa, fragB_row(vtt + 32 * 32, 32, 0), o2);
    o3 = WMMA_BF16(pa, fragB_row(vtt + 48 * 32, 32, 0), o3);
    __syncthreads();
  }
  int b = bh >> 4, h = bh & 15;
  int half = lane >> 4, cn = lane & 15;
#pragma unroll
  for (int i = 0; i < 8; ++i) {
    int s = mrow + i + 8 * half;
    size_t base = ((size_t)(b * 1024 + s)) * 1024 + h * 64;
    float inv = 1.f / lsum[i];
    ao[base + cn]      = f2bf(o0[i] * inv);
    ao[base + 16 + cn] = f2bf(o1[i] * inv);
    ao[base + 32 + cn] = f2bf(o2[i] * inv);
    ao[base + 48 + cn] = f2bf(o3[i] * inv);
  }
}

// ---------------------------------------------------------------------------
// x2 = x + ao @ w_o^T + b_o ; also initializes out = x2.
__global__ __launch_bounds__(256) void k_gemm_wo(const unsigned short* __restrict__ ao,
                                                 const float* __restrict__ w,
                                                 const float* __restrict__ bias,
                                                 const float* __restrict__ xin,
                                                 float* __restrict__ x2,
                                                 float* __restrict__ out) {
  __shared__ __align__(16) unsigned short At[64 * 32];
  __shared__ __align__(16) float          Bf[64 * 32];
  __shared__ __align__(16) unsigned short Bt[64 * 32];
  int m0 = blockIdx.y * 64, n0 = blockIdx.x * 64;
  v8f acc0 = {}, acc1 = {};
  gemm_mainloop(ao, 1024, w, 1024, 1024, m0, n0, At, Bf, Bt, acc0, acc1);

  int wv = threadIdx.x >> 5, lane = threadIdx.x & 31;
  int mt = (wv & 3) * 16, nt = (wv >> 2) * 32;
  int half = lane >> 4, cn = lane & 15;
#pragma unroll
  for (int i = 0; i < 8; ++i) {
    int tok = m0 + mt + i + 8 * half;
    int na = n0 + nt + cn, nb = na + 16;
    size_t oa = (size_t)tok * 1024 + na, ob = (size_t)tok * 1024 + nb;
    float va = acc0[i] + bias[na] + xin[oa];
    float vb = acc1[i] + bias[nb] + xin[ob];
    x2[oa] = va; out[oa] = va;
    x2[ob] = vb; out[ob] = vb;
  }
}

// ---------------------------------------------------------------------------
// Gate: scores -> top-6 -> softmax -> dense comb[N,16] (zeros elsewhere).
__global__ __launch_bounds__(256) void k_gate(const unsigned short* __restrict__ h2,
                                              const float* __restrict__ gw,
                                              const float* __restrict__ gb,
                                              float* __restrict__ comb) {
  int tok = blockIdx.x * 8 + (threadIdx.x >> 5);
  int lane = threadIdx.x & 31;
  const unsigned short* hr = h2 + (size_t)tok * 1024;
  float sc[16];
#pragma unroll 1
  for (int e = 0; e < 16; ++e) {
    const float* wr = gw + (size_t)e * 1024;
    float p = 0.f;
    for (int d = lane; d < 1024; d += 32) p += bf2f(hr[d]) * wr[d];
#pragma unroll
    for (int o = 16; o; o >>= 1) p += __shfl_xor(p, o, 32);
    sc[e] = p + gb[e];
  }
  float tmp[16];
#pragma unroll
  for (int e = 0; e < 16; ++e) tmp[e] = sc[e];
  int bidx[6]; float bval[6];
#pragma unroll
  for (int k = 0; k < 6; ++k) {
    int bi = 0; float bv = tmp[0];
#pragma unroll
    for (int e = 1; e < 16; ++e) if (tmp[e] > bv) { bv = tmp[e]; bi = e; }
    bidx[k] = bi; bval[k] = bv; tmp[bi] = -3e38f;
  }
  float den = 0.f, wk[6];
#pragma unroll
  for (int k = 0; k < 6; ++k) { wk[k] = __expf(bval[k] - bval[0]); den += wk[k]; }
  if (lane == 0) {
    float cb[16];
#pragma unroll
    for (int e = 0; e < 16; ++e) cb[e] = 0.f;
#pragma unroll
    for (int k = 0; k < 6; ++k) cb[bidx[k]] = wk[k] / den;
#pragma unroll
    for (int e = 0; e < 16; ++e) comb[(size_t)tok * 16 + e] = cb[e];
  }
}

// ---------------------------------------------------------------------------
// u = gelu(h2 @ W1_e^T + b1_e); skips 64-token tiles with all comb==0.
__global__ __launch_bounds__(256) void k_ffn1(const unsigned short* __restrict__ h2,
                                              const float* __restrict__ W1e,
                                              const float* __restrict__ b1e,
                                              const float* __restrict__ comb, int e,
                                              unsigned short* __restrict__ u) {
  __shared__ int active;
  __shared__ __align__(16) unsigned short At[64 * 32];
  __shared__ __align__(16) float          Bf[64 * 32];
  __shared__ __align__(16) unsigned short Bt[64 * 32];
  int m0 = blockIdx.y * 64, n0 = blockIdx.x * 64;
  if (threadIdx.x == 0) active = 0;
  __syncthreads();
  if (threadIdx.x < 64 && comb[(size_t)(m0 + threadIdx.x) * 16 + e] != 0.f) active = 1;
  __syncthreads();
  if (!active) return;
  v8f acc0 = {}, acc1 = {};
  gemm_mainloop(h2, 1024, W1e, 1024, 1024, m0, n0, At, Bf, Bt, acc0, acc1);

  int wv = threadIdx.x >> 5, lane = threadIdx.x & 31;
  int mt = (wv & 3) * 16, nt = (wv >> 2) * 32;
  int half = lane >> 4, cn = lane & 15;
#pragma unroll
  for (int i = 0; i < 8; ++i) {
    int tok = m0 + mt + i + 8 * half;
#pragma unroll
    for (int t = 0; t < 2; ++t) {
      int n = n0 + nt + t * 16 + cn;
      float v = (t ? acc1[i] : acc0[i]) + b1e[n];
      float gl = 0.5f * v * (1.f + erff(v * 0.70710678118654752f));  // exact gelu
      u[(size_t)tok * 4096 + n] = f2bf(gl);
    }
  }
}

// ---------------------------------------------------------------------------
// out += comb[:,e] * (u @ W2_e^T + b2_e); skips inactive tiles.
__global__ __launch_bounds__(256) void k_ffn2(const unsigned short* __restrict__ u,
                                              const float* __restrict__ W2e,
                                              const float* __restrict__ b2e,
                                              const float* __restrict__ comb, int e,
                                              float* __restrict__ out) {
  __shared__ int active;
  __shared__ __align__(16) unsigned short At[64 * 32];
  __shared__ __align__(16) float          Bf[64 * 32];
  __shared__ __align__(16) unsigned short Bt[64 * 32];
  int m0 = blockIdx.y * 64, n0 = blockIdx.x * 64;
  if (threadIdx.x == 0) active = 0;
  __syncthreads();
  if (threadIdx.x < 64 && comb[(size_t)(m0 + threadIdx.x) * 16 + e] != 0.f) active = 1;
  __syncthreads();
  if (!active) return;
  v8f acc0 = {}, acc1 = {};
  gemm_mainloop(u, 4096, W2e, 4096, 4096, m0, n0, At, Bf, Bt, acc0, acc1);

  int wv = threadIdx.x >> 5, lane = threadIdx.x & 31;
  int mt = (wv & 3) * 16, nt = (wv >> 2) * 32;
  int half = lane >> 4, cn = lane & 15;
#pragma unroll
  for (int i = 0; i < 8; ++i) {
    int tok = m0 + mt + i + 8 * half;
    float cw = comb[(size_t)tok * 16 + e];
#pragma unroll
    for (int t = 0; t < 2; ++t) {
      int n = n0 + nt + t * 16 + cn;
      float y = (t ? acc1[i] : acc0[i]) + b2e[n];
      out[(size_t)tok * 1024 + n] += cw * y;
    }
  }
}

// ---------------------------------------------------------------------------
extern "C" void kernel_launch(void* const* d_in, const int* in_sizes, int n_in,
                              void* d_out, int out_size, void* d_ws, size_t ws_size,
                              hipStream_t stream) {
  const float* x      = (const float*)d_in[0];
  const float* ln1_g  = (const float*)d_in[1];
  const float* ln1_b  = (const float*)d_in[2];
  const float* ln2_g  = (const float*)d_in[3];
  const float* ln2_b  = (const float*)d_in[4];
  const float* w_qkv  = (const float*)d_in[5];
  const float* b_qkv  = (const float*)d_in[6];
  const float* w_o    = (const float*)d_in[7];
  const float* b_o    = (const float*)d_in[8];
  const float* gate_w = (const float*)d_in[9];
  const float* gate_b = (const float*)d_in[10];
  const float* W1     = (const float*)d_in[11];
  const float* b1     = (const float*)d_in[12];
  const float* W2     = (const float*)d_in[13];
  const float* b2     = (const float*)d_in[14];
  float* out = (float*)d_out;

  char* wsb = (char*)d_ws;
  const size_t MiB = (size_t)1 << 20;
  unsigned short* h1 = (unsigned short*)(wsb + 0);        // 4 MiB  [2048,1024] bf16
  unsigned short* qh = (unsigned short*)(wsb + 4  * MiB); // 4 MiB  [32,1024,64] bf16
  unsigned short* kh = (unsigned short*)(wsb + 8  * MiB);
  unsigned short* vh = (unsigned short*)(wsb + 12 * MiB);
  unsigned short* ao = (unsigned short*)(wsb + 16 * MiB); // 4 MiB
  float*          x2 = (float*)        (wsb + 20 * MiB);  // 8 MiB f32
  unsigned short* h2 = (unsigned short*)(wsb + 28 * MiB); // 4 MiB
  float*        comb = (float*)        (wsb + 32 * MiB);  // 128 KiB
  unsigned short* u  = (unsigned short*)(wsb + 33 * MiB); // 16 MiB [2048,4096] bf16

  dim3 blk(256);
  k_layernorm<<<2048, blk, 0, stream>>>(x, ln1_g, ln1_b, h1);
  k_gemm_qkv<<<dim3(48, 32), blk, 0, stream>>>(h1, w_qkv, b_qkv, qh, kh, vh);
  k_attn<<<dim3(8, 32), blk, 0, stream>>>(qh, kh, vh, ao);
  k_gemm_wo<<<dim3(16, 32), blk, 0, stream>>>(ao, w_o, b_o, x, x2, out);
  k_layernorm<<<2048, blk, 0, stream>>>(x2, ln2_g, ln2_b, h2);
  k_gate<<<256, blk, 0, stream>>>(h2, gate_w, gate_b, comb);
  for (int e = 0; e < 16; ++e) {
    k_ffn1<<<dim3(64, 32), blk, 0, stream>>>(h2, W1 + (size_t)e * 4096 * 1024,
                                             b1 + (size_t)e * 4096, comb, e, u);
    k_ffn2<<<dim3(16, 32), blk, 0, stream>>>(u, W2 + (size_t)e * 1024 * 4096,
                                             b2 + (size_t)e * 1024, comb, e, out);
  }
  (void)in_sizes; (void)n_in; (void)out_size; (void)ws_size;
}
